// PatchClassicalGenerator_16664473109117
// MI455X (gfx1250) — matrix operand order repaired and checked by reading, not compiled, not run on hardware
//
#include <hip/hip_runtime.h>
#include <math.h>

// Problem constants (from reference)
#define G_N   256      // generators
#define BATCH 1024
#define NDIM  100      // noise dim (padded to 128 for K)
#define KPAD  128
#define HID   128
#define POUT  256      // PATCH*PATCH
#define BM    128      // batch rows per block
#define LDA   136      // LDS row stride (halfs) for A tile  (272B = 17*16B aligned)
#define LDB   136      // LDS row stride (halfs) for W^T tile

typedef __attribute__((ext_vector_type(16))) _Float16 v16h;
typedef __attribute__((ext_vector_type(8)))  _Float16 v8h;
typedef __attribute__((ext_vector_type(8)))  float    v8f;

// A fragment (16x32 f16): lane holds row M=lane&15.
// lanes 0-15 : K = kb+0..7   and kb+16..23
// lanes 16-31: K = kb+8..15  and kb+24..31   (caller adds the +8 half offset)
__device__ __forceinline__ v16h ldA(const _Float16* p) {
    v8h lo = *(const v8h*)(p);        // K +0..7   (16B aligned)
    v8h hi = *(const v8h*)(p + 16);   // K +16..23 (16B aligned)
    v16h a;
#pragma unroll
    for (int i = 0; i < 8; ++i) { a[i] = lo[i]; a[i + 8] = hi[i]; }
    return a;
}

// B fragment (32x16 f16): lane holds column N=lane&15; same K order as A
// (lanes 0-15: K=kb+0..7,16..23, lanes 16-31: K=kb+8..15,24..31; caller adds +8 halfs).
__device__ __forceinline__ v16h ldB(const _Float16* p) {
    v8h lo = *(const v8h*)(p);        // K +0..7
    v8h hi = *(const v8h*)(p + 16);   // K +16..23
    v16h b;
#pragma unroll
    for (int i = 0; i < 8; ++i) { b[i] = lo[i]; b[i + 8] = hi[i]; }
    return b;
}

__device__ __forceinline__ float fast_tanh(float x) {
#if __has_builtin(__builtin_amdgcn_tanhf)
    return __builtin_amdgcn_tanhf(x);   // V_TANH_F32 (CDNA5 trans op)
#elif __has_builtin(__builtin_amdgcn_tanh_f32)
    return __builtin_amdgcn_tanh_f32(x);
#else
    return tanhf(x);
#endif
}

__global__ __launch_bounds__(256)
void patch_gen_fused(const float* __restrict__ noise,
                     const float* __restrict__ W1,
                     const float* __restrict__ b1,
                     const float* __restrict__ W2,
                     const float* __restrict__ b2,
                     float* __restrict__ out)
{
    extern __shared__ char smem[];
    _Float16* As = (_Float16*)smem;                               // [BM][LDA]  (noise, then hidden)
    _Float16* Bt = (_Float16*)(smem + BM * LDA * 2);              // [POUT][LDB] weights transposed
    float*    bS = (float*)(smem + BM * LDA * 2 + POUT * LDB * 2);// [256] bias

    const int tid  = threadIdx.x;
    const int lane = tid & 31;
    const int wv   = tid >> 5;              // wave 0..7 -> 16-row strip
    const int g    = blockIdx.x >> 3;       // generator
    const int row0 = (blockIdx.x & 7) * BM; // batch tile origin

    const int ln  = lane & 15;
    const int hi8  = (lane >> 4) ? 8  : 0;  // A k-offset (halfs) / C row offset
    const int hi16 = (lane >> 4) ? 16 : 0;  // unused; B k-offset is hi8 too

    // ---------------- stage 0: stage noise tile + W1^T in LDS (f32 -> f16) ----
    const float* np = noise + (size_t)g * BATCH * NDIM + (size_t)row0 * NDIM;
#pragma unroll 1
    for (int idx = tid; idx < BM * KPAD; idx += 256) {
        int r = idx >> 7, k = idx & 127;
        float v = (k < NDIM) ? np[r * NDIM + k] : 0.0f;   // zero-pad K 100..127
        As[r * LDA + k] = (_Float16)v;
    }
    const float* w1p = W1 + (size_t)g * NDIM * HID;       // [k][h] row-major
#pragma unroll 1
    for (int idx = tid; idx < NDIM * HID; idx += 256) {   // coalesced in h
        int k = idx >> 7, h = idx & 127;
        Bt[h * LDB + k] = (_Float16)w1p[idx];             // transpose into [h][k]
    }
#pragma unroll 1
    for (int idx = tid; idx < (KPAD - NDIM) * HID; idx += 256) {
        int k = NDIM + (idx >> 7), h = idx & 127;
        Bt[h * LDB + k] = (_Float16)0.0f;
    }
    if (tid < HID) bS[tid] = b1[g * HID + tid];
    __syncthreads();

    // ---------------- stage 1: h = relu(noise @ W1 + b1) -----------------------
    v16h a1[4];
#pragma unroll
    for (int kc = 0; kc < 4; ++kc)
        a1[kc] = ldA(As + (wv * 16 + ln) * LDA + kc * 32 + hi8);

    float hreg[8][8];                                      // [n-tile][acc vgpr]
#pragma unroll
    for (int nt = 0; nt < 8; ++nt) {
        v8f acc = {};
#pragma unroll
        for (int kc = 0; kc < 4; ++kc) {
            v16h b = ldB(Bt + (nt * 16 + ln) * LDB + kc * 32 + hi8);
            acc = __builtin_amdgcn_wmma_f32_16x16x32_f16(
                      false, a1[kc], false, b, (short)0, acc, false, false);
        }
        float bias = bS[nt * 16 + ln];
#pragma unroll
        for (int v = 0; v < 8; ++v)
            hreg[nt][v] = fmaxf(acc[v] + bias, 0.0f);      // bias + ReLU
    }
    __syncthreads();   // everyone done reading As (noise) and Bt (W1)

    // hidden activation becomes the next A matrix (f16, same LDS region)
#pragma unroll
    for (int nt = 0; nt < 8; ++nt)
#pragma unroll
        for (int v = 0; v < 8; ++v)
            As[(wv * 16 + v + hi8) * LDA + nt * 16 + ln] = (_Float16)hreg[nt][v];

    // W2^T into Bt: Bt[p][k] = W2[g][k][p]
    const float* w2p = W2 + (size_t)g * HID * POUT;
#pragma unroll 1
    for (int idx = tid; idx < HID * POUT; idx += 256) {    // coalesced in p
        int k = idx >> 8, p = idx & 255;
        Bt[p * LDB + k] = (_Float16)w2p[idx];
    }
    bS[tid] = b2[g * POUT + tid];
    __syncthreads();

    // ---------------- stage 2: out = tanh(h @ W2 + b2), fused transpose --------
    v16h a2[4];
#pragma unroll
    for (int kc = 0; kc < 4; ++kc)
        a2[kc] = ldA(As + (wv * 16 + ln) * LDA + kc * 32 + hi8);

    // out[b, g, p] at b*(G_N*POUT) + g*POUT + p  (== [B,1,256,256] view)
    float* op = out + (size_t)(row0 + wv * 16) * (G_N * POUT) + g * POUT;
#pragma unroll 1
    for (int nt = 0; nt < 16; ++nt) {
        v8f acc = {};
#pragma unroll
        for (int kc = 0; kc < 4; ++kc) {
            v16h b = ldB(Bt + (nt * 16 + ln) * LDB + kc * 32 + hi8);
            acc = __builtin_amdgcn_wmma_f32_16x16x32_f16(
                      false, a2[kc], false, b, (short)0, acc, false, false);
        }
        float bias = bS[nt * 16 + ln];
#pragma unroll
        for (int v = 0; v < 8; ++v) {
            float x = acc[v] + bias;
            op[(size_t)(v + hi8) * (G_N * POUT) + nt * 16 + ln] = fast_tanh(x);
        }
    }
}

extern "C" void kernel_launch(void* const* d_in, const int* in_sizes, int n_in,
                              void* d_out, int out_size, void* d_ws, size_t ws_size,
                              hipStream_t stream) {
    (void)in_sizes; (void)n_in; (void)out_size; (void)d_ws; (void)ws_size;
    const float* noise = (const float*)d_in[0];
    const float* W1    = (const float*)d_in[1];
    const float* b1    = (const float*)d_in[2];
    const float* W2    = (const float*)d_in[3];
    const float* b2    = (const float*)d_in[4];
    float* out = (float*)d_out;

    const size_t smem = (size_t)BM * LDA * 2     // A tile      (34,816 B)
                      + (size_t)POUT * LDB * 2   // W^T tile    (69,632 B)
                      + 256 * sizeof(float);     // bias        ( 1,024 B)  -> ~103 KB

    dim3 grid(G_N * (BATCH / BM));               // 2048 blocks
    dim3 block(256);                             // 8 wave32s
    patch_gen_fused<<<grid, block, smem, stream>>>(noise, W1, b1, W2, b2, out);
}
